// AttnBlock_19980187861277
// MI455X (gfx1250) — matrix operand order repaired and checked by reading, not compiled, not run on hardware
//
#include <hip/hip_runtime.h>
#include <hip/hip_bf16.h>
#include <math.h>

// ---------------------------------------------------------------------------
// AttnBlock for MI455X (gfx1250, wave32, WMMA + Tensor Data Mover).
// GEMMs: v_wmma_f32_16x16x32_f16, 128x128 workgroup tiles staged in LDS by
// the TDM (tensor_load_to_lds, TENSORcnt double-buffering), 64x32 wave tiles.
// "BT" convention: B operand stored transposed, element (k, col) at
// BT[col*ldb + k], so LDS tile rows are contiguous K-slabs for both operands.
// ---------------------------------------------------------------------------

typedef __attribute__((ext_vector_type(16))) _Float16 v16h;
typedef __attribute__((ext_vector_type(8)))  _Float16 v8h;
typedef __attribute__((ext_vector_type(8)))  float    v8f;

#define BATCH 4
#define CCH   512          // channels
#define NPIX  4096         // H*W
#define NGRP  32
#define CPG   16           // channels per group

#define LDSPITCH 40        // halves per LDS tile row (32 data + 8 pad -> 80B)

#if defined(__has_builtin)
#if __has_builtin(__builtin_amdgcn_tensor_load_to_lds) && \
    __has_builtin(__builtin_amdgcn_s_wait_tensorcnt)
#define USE_TDM 1
#endif
#endif

// ---------------------------------------------------------------------------
// TDM descriptor build + issue: 2-D tile, 32 halves (dim0) x 128 rows (dim1),
// data_size = 2 bytes, row stride = ld elements, LDS padding 4 DWORDs every
// 16 DWORDs (-> LDSPITCH-half rows in LDS).  ISA 8.3/8.4 bitfields.
// 6-arg builtin form (clang-23 lane): (v4u, v8i, v4i, v4i, v8i, i32 cpol).
// ---------------------------------------------------------------------------
#ifdef USE_TDM
__device__ __forceinline__ void tdm_load_tile(unsigned lds_off,
                                              const _Float16* gptr, int ld) {
  typedef unsigned int v4u __attribute__((ext_vector_type(4)));
  typedef int v8i_ __attribute__((ext_vector_type(8)));
  typedef int v4i_ __attribute__((ext_vector_type(4)));
  unsigned long long ga = (unsigned long long)(uintptr_t)gptr;
  v4u g0;
  g0[0] = 1u;                                         // count=1, user mode
  g0[1] = lds_off;                                    // lds_addr
  g0[2] = (unsigned)(ga & 0xffffffffu);               // global_addr lo
  g0[3] = (unsigned)((ga >> 32) & 0x01ffffffu)        // global_addr hi (57b)
          | (2u << 30);                               // type=2 ("image")
  unsigned td0 = (unsigned)ld;                        // tensor_dim0
  unsigned td1 = 128u;                                // tensor_dim1
  unsigned long long st = (unsigned long long)ld;     // tensor_dim0_stride
  v8i_ g1;
  g1[0] = (1 << 16)                                   // data_size = 2 bytes
        | (1 << 20)                                   // pad_enable
        | (3 << 22)                                   // pad_interval: 16 DW
        | (3 << 25);                                  // pad_amount: 4 DW
  g1[1] = (int)((td0 & 0xffffu) << 16);               // tensor_dim0 [63:48]
  g1[2] = (int)((td0 >> 16) | ((td1 & 0xffffu) << 16));
  g1[3] = (int)((td1 >> 16) | (32u << 16));           // tile_dim0 = 32 halves
  g1[4] = (int)128;                                   // tile_dim1 = 128 rows
  g1[5] = (int)(st & 0xffffffffu);                    // stride [191:160]
  g1[6] = (int)((st >> 32) & 0xffffu);                // stride [207:192]
  g1[7] = 0;
  v4i_ z4 = {0, 0, 0, 0};
  v8i_ z8 = {0, 0, 0, 0, 0, 0, 0, 0};
  __builtin_amdgcn_tensor_load_to_lds(g0, g1, z4, z4, z8, 0);
}
#endif

// ---------------------------------------------------------------------------
// Fragment loaders from LDS tiles (16-bit WMMA operand layouts, ISA 7.12.2).
// Tile row r holds 32 contiguous K-halves at pitch LDSPITCH.
// ---------------------------------------------------------------------------
__device__ __forceinline__ v16h frag_a_lds(const _Float16* L, int r0, int lane) {
  int r  = lane & 15;
  int ko = (lane & 16) ? 8 : 0;
  const _Float16* p = L + (r0 + r) * LDSPITCH + ko;
  v8h lo = *(const v8h*)(p);
  v8h hi = *(const v8h*)(p + 16);
  v16h f;
#pragma unroll
  for (int i = 0; i < 8; ++i) { f[i] = lo[i]; f[i + 8] = hi[i]; }
  return f;
}

__device__ __forceinline__ v16h frag_b_lds(const _Float16* L, int c0, int lane) {
  int c  = lane & 15;
  int ko = (lane & 16) ? 8 : 0;
  const _Float16* p = L + (c0 + c) * LDSPITCH + ko;
  v8h lo = *(const v8h*)(p);
  v8h hi = *(const v8h*)(p + 16);
  v16h f;
#pragma unroll
  for (int i = 0; i < 8; ++i) { f[i] = lo[i]; f[i + 8] = hi[i]; }
  return f;
}

// ---------------------------------------------------------------------------
// Generic WMMA GEMM: D[M,N] = A[M,K] * BT[N,K]^T.
// Block = 256 threads = 8 waves (2 M x 4 N), wave tile 64x32, WG tile 128x128.
// Grid: (M/128, N/128, batch).
// mode 0: Dh[m*ldd+n] = f16((acc+bias)*scale)            (natural)
// mode 1: Dh[n*ldd+m] = f16((acc+bias)*scale)            (transposed)
// mode 2: Df[n*ldd+m] = resid[n*ldd+m] + acc + bias      (f32 + residual)
// ---------------------------------------------------------------------------
__global__ __launch_bounds__(256) void k_gemm(
    const _Float16* __restrict__ A,  long long strideA, int lda,
    const _Float16* __restrict__ BT, long long strideB, int ldb,
    int Kdim, int mode, float out_scale,
    const float* __restrict__ bias_m, const float* __restrict__ bias_n,
    const float* __restrict__ resid,
    _Float16* __restrict__ Dh, float* __restrict__ Df,
    long long strideD, int ldd) {
  __shared__ _Float16 smem[2][2][128 * LDSPITCH];   // [buf][A|B][tile] = 40 KB

  const int tid  = threadIdx.x;
  const int lane = tid & 31;
  const int wave = tid >> 5;
  const int wm = wave >> 2;                     // 0..1
  const int wn = wave & 3;                      // 0..3
  const int rowg = blockIdx.x * 128;            // WG M origin
  const int colg = blockIdx.y * 128;            // WG N origin
  const int z = blockIdx.z;

  const _Float16* Az = A  + (size_t)z * (size_t)strideA;
  const _Float16* Bz = BT + (size_t)z * (size_t)strideB;

  const int nsteps = Kdim / 32;

#ifdef USE_TDM
  if (wave == 0) {   // wave 0 drives the Tensor Data Mover
    tdm_load_tile((unsigned)(uintptr_t)&smem[0][0][0],
                  Az + (size_t)rowg * lda, lda);
    tdm_load_tile((unsigned)(uintptr_t)&smem[0][1][0],
                  Bz + (size_t)colg * ldb, ldb);
  }
#else
  // Cooperative fallback copy of k-step 0 into buffer 0.
  {
#pragma unroll
    for (int it = 0; it < 2; ++it) {
      int idx = tid + it * 256;                 // 0..511
      int row = idx >> 2;                       // 0..127
      int off = (idx & 3) * 8;                  // 0,8,16,24
      *(v8h*)(&smem[0][0][0] + row * LDSPITCH + off) =
          *(const v8h*)(Az + (size_t)(rowg + row) * lda + off);
      *(v8h*)(&smem[0][1][0] + row * LDSPITCH + off) =
          *(const v8h*)(Bz + (size_t)(colg + row) * ldb + off);
    }
  }
  __syncthreads();
#endif

  v8f acc[4][2] = {};

  for (int s = 0; s < nsteps; ++s) {
    const int buf = s & 1;
#ifdef USE_TDM
    if (wave == 0) {
      if (s + 1 < nsteps) {
        const int k1 = (s + 1) * 32;
        tdm_load_tile((unsigned)(uintptr_t)&smem[buf ^ 1][0][0],
                      Az + (size_t)rowg * lda + k1, lda);
        tdm_load_tile((unsigned)(uintptr_t)&smem[buf ^ 1][1][0],
                      Bz + (size_t)colg * ldb + k1, ldb);
        __builtin_amdgcn_s_wait_tensorcnt(2);   // current pair complete
      } else {
        __builtin_amdgcn_s_wait_tensorcnt(0);
      }
    }
    __syncthreads();                            // tiles visible to all waves
#else
    if (s + 1 < nsteps) {                       // overlap copy of next slab
      const int k1 = (s + 1) * 32;
#pragma unroll
      for (int it = 0; it < 2; ++it) {
        int idx = tid + it * 256;
        int row = idx >> 2;
        int off = (idx & 3) * 8;
        *(v8h*)(&smem[buf ^ 1][0][0] + row * LDSPITCH + off) =
            *(const v8h*)(Az + (size_t)(rowg + row) * lda + k1 + off);
        *(v8h*)(&smem[buf ^ 1][1][0] + row * LDSPITCH + off) =
            *(const v8h*)(Bz + (size_t)(colg + row) * ldb + k1 + off);
      }
    }
#endif
    const _Float16* LA = &smem[buf][0][0];
    const _Float16* LB = &smem[buf][1][0];
    v16h af[4], bf[2];
#pragma unroll
    for (int mi = 0; mi < 4; ++mi) af[mi] = frag_a_lds(LA, wm * 64 + mi * 16, lane);
#pragma unroll
    for (int ni = 0; ni < 2; ++ni) bf[ni] = frag_b_lds(LB, wn * 32 + ni * 16, lane);
#pragma unroll
    for (int mi = 0; mi < 4; ++mi) {
#pragma unroll
      for (int ni = 0; ni < 2; ++ni) {
        acc[mi][ni] = __builtin_amdgcn_wmma_f32_16x16x32_f16(
            false, af[mi], false, bf[ni], (short)0, acc[mi][ni], false, false);
      }
    }
    __syncthreads();                            // done reading buf
  }

  // Epilogue. C/D layout: VGPR r holds (M = r or r+8, N = lane&15).
  const int nl = lane & 15;
  const int mo = (lane & 16) ? 8 : 0;
#pragma unroll
  for (int mi = 0; mi < 4; ++mi) {
#pragma unroll
    for (int ni = 0; ni < 2; ++ni) {
#pragma unroll
      for (int r = 0; r < 8; ++r) {
        int m = rowg + wm * 64 + mi * 16 + mo + r;
        int n = colg + wn * 32 + ni * 16 + nl;
        float v = acc[mi][ni][r];
        if (bias_m) v += bias_m[m];
        if (bias_n) v += bias_n[n];
        v *= out_scale;
        if (mode == 0) {
          Dh[(size_t)z * (size_t)strideD + (size_t)m * ldd + n] = (_Float16)v;
        } else if (mode == 1) {
          Dh[(size_t)z * (size_t)strideD + (size_t)n * ldd + m] = (_Float16)v;
        } else {
          size_t idx = (size_t)z * (size_t)strideD + (size_t)n * ldd + m;
          Df[idx] = resid[idx] + v;
        }
      }
    }
  }
}

// ---------------------------------------------------------------------------
// Weight conversion f32 -> f16 (all four 512x512 matrices kept row-major).
// ---------------------------------------------------------------------------
__global__ void k_cvt_w(const float* __restrict__ a, const float* __restrict__ b,
                        const float* __restrict__ c, const float* __restrict__ d,
                        _Float16* __restrict__ ah, _Float16* __restrict__ bh,
                        _Float16* __restrict__ ch, _Float16* __restrict__ dh) {
  int i = blockIdx.x * blockDim.x + threadIdx.x;
  if (i < CCH * CCH) {
    ah[i] = (_Float16)a[i];
    bh[i] = (_Float16)b[i];
    ch[i] = (_Float16)c[i];
    dh[i] = (_Float16)d[i];
  }
}

// ---------------------------------------------------------------------------
// GroupNorm pass 1: per (batch, group) mean / rsqrt(var+eps).
// Group data is a contiguous run of 16*4096 = 65536 floats.
// ---------------------------------------------------------------------------
__global__ __launch_bounds__(256) void k_gn_stats(const float* __restrict__ x,
                                                  float* __restrict__ mu,
                                                  float* __restrict__ rs) {
  const int g = blockIdx.x, b = blockIdx.y;
  const float4* p4 = (const float4*)(x + ((size_t)(b * CCH + g * CPG)) * NPIX);
  float s = 0.f, ss = 0.f;
  for (int i = threadIdx.x; i < (CPG * NPIX) / 4; i += 256) {
    float4 q = p4[i];
    s  += q.x + q.y + q.z + q.w;
    ss += q.x * q.x + q.y * q.y + q.z * q.z + q.w * q.w;
  }
  __shared__ float r0[256], r1[256];
  r0[threadIdx.x] = s; r1[threadIdx.x] = ss;
  __syncthreads();
  for (int ofs = 128; ofs > 0; ofs >>= 1) {
    if (threadIdx.x < ofs) {
      r0[threadIdx.x] += r0[threadIdx.x + ofs];
      r1[threadIdx.x] += r1[threadIdx.x + ofs];
    }
    __syncthreads();
  }
  if (threadIdx.x == 0) {
    const float inv_n = 1.0f / (float)(CPG * NPIX);
    float m = r0[0] * inv_n;
    float var = r1[0] * inv_n - m * m;
    mu[b * NGRP + g] = m;
    rs[b * NGRP + g] = rsqrtf(var + 1e-5f);
  }
}

// ---------------------------------------------------------------------------
// GroupNorm pass 2 + transpose: xnT[b][n][c] (f16) via 32x32 LDS tile.
// Grid (C/32, N/32, B), 256 threads.
// ---------------------------------------------------------------------------
__global__ __launch_bounds__(256) void k_gn_norm(const float* __restrict__ x,
                                                 const float* __restrict__ gw,
                                                 const float* __restrict__ gb,
                                                 const float* __restrict__ mu,
                                                 const float* __restrict__ rs,
                                                 _Float16* __restrict__ xnT) {
  __shared__ float tile[32][33];
  const int b = blockIdx.z;
  const int c0 = blockIdx.x * 32;
  const int n0 = blockIdx.y * 32;
  const int tx = threadIdx.x & 31, ty = threadIdx.x >> 5;  // ty: 0..7
#pragma unroll
  for (int i = 0; i < 4; ++i) {
    int c = c0 + ty + i * 8;
    tile[ty + i * 8][tx] = x[((size_t)(b * CCH + c)) * NPIX + n0 + tx];
  }
  __syncthreads();
#pragma unroll
  for (int i = 0; i < 4; ++i) {
    int n = n0 + ty + i * 8;
    int c = c0 + tx;
    int gi = b * NGRP + (c >> 4);
    float v = (tile[tx][ty + i * 8] - mu[gi]) * rs[gi] * gw[c] + gb[c];
    xnT[((size_t)(b * NPIX + n)) * CCH + c] = (_Float16)v;
  }
}

// ---------------------------------------------------------------------------
// Row softmax (in place), f16 rows of length 4096. One 256-thread WG per row.
// ---------------------------------------------------------------------------
__global__ __launch_bounds__(256) void k_softmax(_Float16* __restrict__ S) {
  const int row = blockIdx.x;                   // 0 .. BATCH*NPIX-1
  _Float16* p = S + (size_t)row * NPIX;
  const int t = threadIdx.x;
  float v[16];
  float m = -1e30f;
#pragma unroll
  for (int i = 0; i < 16; ++i) {
    v[i] = (float)p[t + i * 256];
    m = fmaxf(m, v[i]);
  }
  __shared__ float red[256];
  red[t] = m;
  __syncthreads();
  for (int ofs = 128; ofs > 0; ofs >>= 1) {
    if (t < ofs) red[t] = fmaxf(red[t], red[t + ofs]);
    __syncthreads();
  }
  m = red[0];
  __syncthreads();
  float s = 0.f;
#pragma unroll
  for (int i = 0; i < 16; ++i) { v[i] = __expf(v[i] - m); s += v[i]; }
  red[t] = s;
  __syncthreads();
  for (int ofs = 128; ofs > 0; ofs >>= 1) {
    if (t < ofs) red[t] += red[t + ofs];
    __syncthreads();
  }
  const float inv = 1.0f / red[0];
#pragma unroll
  for (int i = 0; i < 16; ++i) p[t + i * 256] = (_Float16)(v[i] * inv);
}

// ---------------------------------------------------------------------------
// Host-side orchestration.
// ---------------------------------------------------------------------------
extern "C" void kernel_launch(void* const* d_in, const int* in_sizes, int n_in,
                              void* d_out, int out_size, void* d_ws, size_t ws_size,
                              hipStream_t stream) {
  (void)in_sizes; (void)n_in; (void)out_size; (void)ws_size;

  const float* x    = (const float*)d_in[0];
  const float* gn_w = (const float*)d_in[1];
  const float* gn_b = (const float*)d_in[2];
  const float* wq   = (const float*)d_in[3];
  const float* bq   = (const float*)d_in[4];
  const float* wk   = (const float*)d_in[5];
  const float* bk   = (const float*)d_in[6];
  const float* wv   = (const float*)d_in[7];
  const float* bv   = (const float*)d_in[8];
  const float* wp   = (const float*)d_in[9];
  const float* bp   = (const float*)d_in[10];
  float* out = (float*)d_out;

  // Workspace carve-up (256B aligned).
  char* ws = (char*)d_ws;
  size_t off = 0;
  auto carve = [&](size_t bytes) -> char* {
    char* p = ws + off;
    off = (off + bytes + 255) & ~(size_t)255;
    return p;
  };
  const size_t W_BYTES   = (size_t)CCH * CCH * 2;            // 512 KB each
  const size_t ACT_BYTES = (size_t)BATCH * NPIX * CCH * 2;   // 16 MB each
  const size_t S_BYTES   = (size_t)BATCH * NPIX * NPIX * 2;  // 128 MB

  _Float16* wq_h = (_Float16*)carve(W_BYTES);
  _Float16* wk_h = (_Float16*)carve(W_BYTES);
  _Float16* wv_h = (_Float16*)carve(W_BYTES);
  _Float16* wp_h = (_Float16*)carve(W_BYTES);
  float*    mu   = (float*)carve(BATCH * NGRP * 4);
  float*    rs   = (float*)carve(BATCH * NGRP * 4);
  _Float16* xnT  = (_Float16*)carve(ACT_BYTES);   // [b][n][c]
  _Float16* qT   = (_Float16*)carve(ACT_BYTES);   // [b][n][c]  (scale folded in)
  _Float16* kT   = (_Float16*)carve(ACT_BYTES);   // [b][n][c]
  _Float16* v_   = (_Float16*)carve(ACT_BYTES);   // [b][c][n]
  _Float16* oT   = (_Float16*)carve(ACT_BYTES);   // [b][n][c]
  _Float16* S    = (_Float16*)carve(S_BYTES);     // [b][i][j]

  const float scale = 0.044194173824159216f;      // 512^-0.5
  const long long ACT = (long long)NPIX * CCH;    // per-batch activation elems
  const long long ATT = (long long)NPIX * NPIX;   // per-batch attention elems

  // 1) weights -> f16
  k_cvt_w<<<dim3((CCH * CCH + 255) / 256), 256, 0, stream>>>(
      wq, wk, wv, wp, wq_h, wk_h, wv_h, wp_h);

  // 2) GroupNorm stats + normalize/transpose
  k_gn_stats<<<dim3(NGRP, BATCH), 256, 0, stream>>>(x, mu, rs);
  k_gn_norm<<<dim3(CCH / 32, NPIX / 32, BATCH), 256, 0, stream>>>(
      x, gn_w, gn_b, mu, rs, xnT);

  // 3) QKV projections: D[o][n] = W[o][c] * xn[c][n]   (BT = xnT)
  dim3 gQKV(CCH / 128, NPIX / 128, BATCH);
  // q -> qT[b][n][o], scaled by 1/sqrt(C)
  k_gemm<<<gQKV, 256, 0, stream>>>(wq_h, 0, CCH, xnT, ACT, CCH, CCH,
                                   1, scale, bq, nullptr, nullptr,
                                   qT, nullptr, ACT, CCH);
  // k -> kT[b][n][o]
  k_gemm<<<gQKV, 256, 0, stream>>>(wk_h, 0, CCH, xnT, ACT, CCH, CCH,
                                   1, 1.0f, bk, nullptr, nullptr,
                                   kT, nullptr, ACT, CCH);
  // v -> v[b][o][n] (natural)
  k_gemm<<<gQKV, 256, 0, stream>>>(wv_h, 0, CCH, xnT, ACT, CCH, CCH,
                                   0, 1.0f, bv, nullptr, nullptr,
                                   v_, nullptr, ACT, NPIX);

  // 4) S[i][j] = sum_c qT[i][c] * kT[j][c]   (A = qT, BT = kT), M=N=4096, K=512
  dim3 gS(NPIX / 128, NPIX / 128, BATCH);
  k_gemm<<<gS, 256, 0, stream>>>(qT, ACT, CCH, kT, ACT, CCH, CCH,
                                 0, 1.0f, nullptr, nullptr, nullptr,
                                 S, nullptr, ATT, NPIX);

  // 5) softmax rows
  k_softmax<<<dim3(BATCH * NPIX), 256, 0, stream>>>(S);

  // 6) oT[i][c] = sum_j P[i][j] * v[c][j]   (A = P, BT = v), M=4096, N=512, K=4096
  dim3 gO(NPIX / 128, CCH / 128, BATCH);
  k_gemm<<<gO, 256, 0, stream>>>(S, ATT, NPIX, v_, ACT, NPIX, NPIX,
                                 0, 1.0f, nullptr, nullptr, nullptr,
                                 oT, nullptr, ACT, CCH);

  // 7) out[b][o][n] = x + bp[o] + sum_c oT[n][c] * wp[o][c]
  //    (A = oT: M=n, K=c; BT = wp_h: col=o), mode 2 writes f32 + residual.
  dim3 gP(NPIX / 128, CCH / 128, BATCH);
  k_gemm<<<gP, 256, 0, stream>>>(oT, ACT, CCH, wp_h, 0, CCH, CCH,
                                 2, 1.0f, nullptr, bp, x,
                                 nullptr, out, ACT, NPIX);
}
